// Head_49263274885985
// MI455X (gfx1250) — matrix operand order repaired and checked by reading, not compiled, not run on hardware
//
#include <hip/hip_runtime.h>

#define EMB   1024
#define HEAD  64
#define BATCH 4
#define SEQ   4096
#define NROW  (BATCH * SEQ)   // 16384 total rows

typedef __attribute__((ext_vector_type(16))) __bf16 v16bf;
typedef __attribute__((ext_vector_type(8)))  float  v8f;

// native f32 -> bf16 (backend emits v_cvt bf16 ops, RTNE)
static __device__ __forceinline__ __bf16 f2bf(float f) {
  return static_cast<__bf16>(f);
}

static __device__ __forceinline__ v8f wmma_bf16(v16bf a, v16bf b, v8f c) {
  // D(16x16 f32) = A(16x32 bf16) * B(32x16 bf16) + C
  return __builtin_amdgcn_wmma_f32_16x16x32_bf16(false, a, false, b, (short)0, c,
                                                 false, false);
}

// ---------------------------------------------------------------------------
// Kernel 0: convert W{q,k,v} fp32 -> bf16, swizzled into B-fragment order:
//   Wb[((mtx*32 + kblk)*64 + n)*32 + (k & 31)]
// so a WMMA B fragment (k in [kblk*32, kblk*32+32), col n, half hi) is the
// 16 contiguous bf16 at offset ((mtx*32+kblk)*64 + n)*32 + hi*16.
// ---------------------------------------------------------------------------
__global__ __launch_bounds__(256)
void wconv(const float* __restrict__ Wq, const float* __restrict__ Wk,
           const float* __restrict__ Wv, __bf16* __restrict__ Wb)
{
  const int i = blockIdx.x * 256 + threadIdx.x;      // 3*1024*64 = 196608 total
  const int mtx = i >> 16;                           // / 65536
  const int rem = i & 65535;
  const int k   = rem >> 6;                          // row of W (EMB dim)
  const int n   = rem & 63;                          // col of W (HEAD dim)
  const float* W = (mtx == 0) ? Wq : (mtx == 1) ? Wk : Wv;
  const float scale = (mtx == 0) ? 0.03125f : 1.0f;  // fold 1/sqrt(1024) into Wq
  const int kblk = k >> 5, kin = k & 31;
  Wb[(((size_t)mtx * 32 + kblk) * 64 + n) * 32 + kin] = f2bf(W[k * 64 + n] * scale);
}

// ---------------------------------------------------------------------------
// Kernel 1: fused QKV projection. One wave computes a 16-row x 64-col tile of
// Q, K AND V: the x A-fragment is loaded/converted once per k-step and feeds
// 12 WMMAs. B fragments are contiguous 32B vector loads from pre-swizzled Wb.
// V is stored transposed [b][head][seq].
// ---------------------------------------------------------------------------
__global__ __launch_bounds__(128)
void qkv_proj(const float* __restrict__ x,
              const __bf16* __restrict__ Wb,
              __bf16* __restrict__ Qb,
              __bf16* __restrict__ Kb,
              __bf16* __restrict__ Vt)
{
  const int w    = threadIdx.x >> 5;
  const int lane = threadIdx.x & 31;
  const int hi   = lane >> 4;     // half-wave select
  const int lo   = lane & 15;
  const int row0 = (blockIdx.x * 4 + w) * 16;

  v8f acc[3][4];
  #pragma unroll
  for (int m = 0; m < 3; ++m)
    #pragma unroll
    for (int nt = 0; nt < 4; ++nt)
      #pragma unroll
      for (int r = 0; r < 8; ++r) acc[m][nt][r] = 0.0f;

  // A-fragment rows: lane lo holds row (row0+lo); k base offset for this half
  const float* __restrict__ xr = x + (size_t)(row0 + lo) * EMB + hi * 16;

  for (int kblk = 0; kblk < 32; ++kblk) {
    const int k0 = kblk * 32;
    v16bf a;
    #pragma unroll
    for (int e = 0; e < 16; ++e) {
      const int kk = e;                               // A layout in-lane k pattern
      a[e] = f2bf(xr[k0 + kk]);
    }
    #pragma unroll
    for (int m = 0; m < 3; ++m) {
      const __bf16* wbase = Wb + (((size_t)m * 32 + kblk) * 64 + lo) * 32 + hi * 16;
      #pragma unroll
      for (int nt = 0; nt < 4; ++nt) {
        v16bf bf = *(const v16bf*)(wbase + (size_t)nt * 16 * 32);
        acc[m][nt] = wmma_bf16(a, bf, acc[m][nt]);
      }
    }
  }

  // stores: Q,K row-major; V transposed (straight-line, no per-store branching)
  const int b  = row0 >> 12;                          // row0 / SEQ
  const int tq0 = row0 & (SEQ - 1);
  #pragma unroll
  for (int nt = 0; nt < 4; ++nt) {
    #pragma unroll
    for (int r = 0; r < 8; ++r) {
      const int m = r + hi * 8;                       // C layout row
      const size_t g = (size_t)(row0 + m);
      Qb[g * HEAD + nt * 16 + lo] = f2bf(acc[0][nt][r]);
      Kb[g * HEAD + nt * 16 + lo] = f2bf(acc[1][nt][r]);
      Vt[((size_t)b * HEAD + nt * 16 + lo) * SEQ + tq0 + m] = f2bf(acc[2][nt][r]);
    }
  }
}

// ---------------------------------------------------------------------------
// Kernel 2: flash attention. One wave owns 16 query rows; streams 32-key
// tiles up to the causal boundary with online softmax; all matmuls on WMMA.
// ---------------------------------------------------------------------------
__global__ __launch_bounds__(128)
void attn_fwd(const __bf16* __restrict__ Qb,
              const __bf16* __restrict__ Kb,
              const __bf16* __restrict__ Vt,
              float* __restrict__ out)
{
  __shared__ __bf16 pls[4][16][32];   // per-wave P transpose staging (C -> A layout)

  const int w    = threadIdx.x >> 5;
  const int lane = threadIdx.x & 31;
  const int hi   = lane >> 4;
  const int lo   = lane & 15;

  const int g  = blockIdx.x * 4 + w;
  const int b  = g >> 8;              // SEQ/16 = 256 query tiles per batch
  const int qt = g & 255;
  const int q0 = qt * 16;
  const size_t base = (size_t)b * SEQ;

  // Q fragment (A layout), 16 rows x 64 deep -> two K=32 fragments, in regs
  v16bf qa[2];
  const __bf16* __restrict__ qrow = Qb + (base + q0 + lo) * HEAD;
  #pragma unroll
  for (int h = 0; h < 2; ++h)
    #pragma unroll
    for (int e = 0; e < 16; ++e) {
      const int kk = h * 32 + hi * 16 + e;
      qa[h][e] = qrow[kk];
    }

  v8f o0, o1, o2, o3;
  #pragma unroll
  for (int r = 0; r < 8; ++r) { o0[r] = 0.f; o1[r] = 0.f; o2[r] = 0.f; o3[r] = 0.f; }
  float mstat[8], lstat[8];
  #pragma unroll
  for (int r = 0; r < 8; ++r) { mstat[r] = -1e30f; lstat[r] = 0.0f; }

  const int jmax = (q0 + 15) >> 5;    // last 32-key tile intersecting causal band
  for (int j = 0; j <= jmax; ++j) {
    const int kb0 = j * 32;

    // ---- S = Q K^T for 32 keys (two 16-key halves); K fragments contiguous ----
    const __bf16* kp0 = Kb + (base + kb0      + lo) * HEAD + hi * 16;
    const __bf16* kp1 = Kb + (base + kb0 + 16 + lo) * HEAD + hi * 16;
    v16bf kf00 = *(const v16bf*)(kp0);
    v16bf kf01 = *(const v16bf*)(kp0 + 32);
    v16bf kf10 = *(const v16bf*)(kp1);
    v16bf kf11 = *(const v16bf*)(kp1 + 32);

    v8f s0, s1;
    #pragma unroll
    for (int r = 0; r < 8; ++r) { s0[r] = 0.0f; s1[r] = 0.0f; }
    s0 = wmma_bf16(qa[0], kf00, s0);
    s0 = wmma_bf16(qa[1], kf01, s0);
    s1 = wmma_bf16(qa[0], kf10, s1);
    s1 = wmma_bf16(qa[1], kf11, s1);

    if (j < jmax) __builtin_prefetch(kp0 + 32 * HEAD, 0, 0);

    // ---- causal mask + online softmax (reductions in 16-lane half-groups) ----
    float p0a[8], p1a[8];
    const int key0 = kb0 + lo, key1 = kb0 + 16 + lo;
    #pragma unroll
    for (int r = 0; r < 8; ++r) {
      const int q = q0 + r + hi * 8;
      float v0 = (key0 <= q) ? s0[r] : -1e30f;
      float v1 = (key1 <= q) ? s1[r] : -1e30f;
      float mx = fmaxf(v0, v1);
      #pragma unroll
      for (int d = 1; d < 16; d <<= 1) mx = fmaxf(mx, __shfl_xor(mx, d, 32));
      const float mnew = fmaxf(mstat[r], mx);
      const float p0 = __expf(v0 - mnew);
      const float p1 = __expf(v1 - mnew);
      float rs = p0 + p1;
      #pragma unroll
      for (int d = 1; d < 16; d <<= 1) rs += __shfl_xor(rs, d, 32);
      const float corr = __expf(mstat[r] - mnew);
      lstat[r] = lstat[r] * corr + rs;
      mstat[r] = mnew;
      o0[r] *= corr; o1[r] *= corr; o2[r] *= corr; o3[r] *= corr;
      p0a[r] = p0; p1a[r] = p1;
    }

    // ---- transpose P from C layout to A layout via per-wave LDS tile ----
    #pragma unroll
    for (int r = 0; r < 8; ++r) {
      const int m = r + hi * 8;
      pls[w][m][lo]      = f2bf(p0a[r]);
      pls[w][m][16 + lo] = f2bf(p1a[r]);
    }
    v16bf pa;                          // LDS ops are in-order within a wave
    #pragma unroll
    for (int e = 0; e < 16; ++e) {
      const int kk = hi * 16 + e;
      pa[e] = pls[w][lo][kk];
    }

    // ---- O += P V ; V^T layout makes B fragments contiguous 32B loads ----
    const __bf16* vb = Vt + ((size_t)b * HEAD + lo) * SEQ + kb0 + hi * 16;
    v16bf vf0 = *(const v16bf*)(vb);
    v16bf vf1 = *(const v16bf*)(vb + 16 * SEQ);
    v16bf vf2 = *(const v16bf*)(vb + 32 * SEQ);
    v16bf vf3 = *(const v16bf*)(vb + 48 * SEQ);
    if (j < jmax) __builtin_prefetch(vb + 32, 0, 0);
    o0 = wmma_bf16(pa, vf0, o0);
    o1 = wmma_bf16(pa, vf1, o1);
    o2 = wmma_bf16(pa, vf2, o2);
    o3 = wmma_bf16(pa, vf3, o3);
  }

  // ---- normalize and store Z (fp32, coalesced across lanes) ----
  #pragma unroll
  for (int r = 0; r < 8; ++r) {
    const int m = r + hi * 8;
    const float inv = 1.0f / lstat[r];
    float* orow = out + (base + q0 + m) * HEAD;
    orow[lo]      = o0[r] * inv;
    orow[16 + lo] = o1[r] * inv;
    orow[32 + lo] = o2[r] * inv;
    orow[48 + lo] = o3[r] * inv;
  }
}

extern "C" void kernel_launch(void* const* d_in, const int* in_sizes, int n_in,
                              void* d_out, int out_size, void* d_ws, size_t ws_size,
                              hipStream_t stream) {
  const float* x  = (const float*)d_in[0];
  const float* Wq = (const float*)d_in[1];
  const float* Wk = (const float*)d_in[2];
  const float* Wv = (const float*)d_in[3];

  // workspace: Q, K row-major bf16; V transposed bf16; swizzled W bf16
  __bf16* Qb = (__bf16*)d_ws;
  __bf16* Kb = Qb + (size_t)NROW * HEAD;
  __bf16* Vt = Kb + (size_t)NROW * HEAD;
  __bf16* Wb = Vt + (size_t)NROW * HEAD;      // 3*1024*64 bf16 = 384 KB

  wconv<<<768, 256, 0, stream>>>(Wq, Wk, Wv, Wb);                 // 196608 elems
  qkv_proj<<<256, 128, 0, stream>>>(x, Wb, Qb, Kb, Vt);           // 1024 row tiles
  attn_fwd<<<256, 128, 0, stream>>>(Qb, Kb, Vt, (float*)d_out);

  (void)in_sizes; (void)n_in; (void)out_size; (void)ws_size;
}
